// AttentionBlock_54391465837005
// MI455X (gfx1250) — compile-verified, matrix operand order checked by reading, not run on hardware
//
#include <hip/hip_runtime.h>
#include <hip/hip_bf16.h>
#include <math.h>

// ---------------- types ----------------
typedef __attribute__((ext_vector_type(16))) __bf16 bf16x16;
typedef __attribute__((ext_vector_type(8)))  float  f32x8;
typedef __attribute__((ext_vector_type(4)))  unsigned int u32x4;
typedef __attribute__((ext_vector_type(8)))  unsigned int u32x8;

#define BQ 2
#define SQ 2048
#define EQ 768
#define HQ 12
#define DQ 64
#define E3 2304
#define E4 3072

__device__ __forceinline__ unsigned short f2bf(float f) {
    union { float f; unsigned u; } x; x.f = f;
    unsigned r = x.u + 0x7FFFu + ((x.u >> 16) & 1u);   // round-to-nearest-even
    return (unsigned short)(r >> 16);
}

__device__ __forceinline__ f32x8 wmma_bf16(bf16x16 a, bf16x16 b, f32x8 c) {
    return __builtin_amdgcn_wmma_f32_16x16x32_bf16(false, a, false, b, (short)0, c, false, false);
}

// Load a 16x32 bf16 A/B fragment from an LDS tile (N-major for B).
__device__ __forceinline__ bf16x16 load_frag(const unsigned short* base, int ldk, int lane) {
    int m = lane & 15, h = (lane >> 4) & 1;
    const unsigned short* p = base + m * ldk + h * 8;
    union { bf16x16 v; uint4 q[2]; } u;
    u.q[0] = *(const uint4*)(p);
    u.q[1] = *(const uint4*)(p + 16);
    return u.v;
}

// ---------------- Tensor Data Mover: 2D tile (64 rows x 64 bf16) -> LDS ----------------
// Builds D# group0/group1 per CDNA5 ISA 08_async_tensor §8.3/8.4 and issues
// tensor_load_to_lds via inline asm (portable across toolchains). Tracked by TENSORcnt.
__device__ __forceinline__ void tdm_load_tile64(unsigned short* lds_dst,
                                                const unsigned short* gsrc,
                                                unsigned k_stride_elems,
                                                unsigned tensor_rows) {
    unsigned long long ga = (unsigned long long)(size_t)gsrc;
    unsigned lds = (unsigned)(size_t)lds_dst;   // LDS aperture: addr[31:0] = LDS byte address
    u32x4 g0;
    g0[0] = 1u;                                         // count=1, user descriptor
    g0[1] = lds;                                        // lds_addr
    g0[2] = (unsigned)ga;                               // global_addr[31:0]
    g0[3] = (unsigned)((ga >> 32) & 0x01FFFFFFu) | 0x80000000u;  // global_addr[56:32] | type=2
    u32x8 g1;
    g1[0] = 1u << 16;                                   // workgroup_mask=0, data_size=1 (2 bytes)
    g1[1] = (k_stride_elems & 0xFFFFu) << 16;           // tensor_dim0[15:0]
    g1[2] = (k_stride_elems >> 16) | ((tensor_rows & 0xFFFFu) << 16);  // dim0 hi | tensor_dim1 lo
    g1[3] = (tensor_rows >> 16) | (64u << 16);          // tensor_dim1 hi | tile_dim0=64
    g1[4] = 64u;                                        // tile_dim1=64, tile_dim2=0
    g1[5] = k_stride_elems;                             // tensor_dim0_stride[31:0] (elements)
    g1[6] = 0u;                                         // stride hi | tensor_dim1_stride lo
    g1[7] = 0u;
    asm volatile("tensor_load_to_lds %0, %1" :: "s"(g0), "s"(g1) : "memory");
}

__device__ __forceinline__ void tdm_wait0() {
    __builtin_amdgcn_s_wait_tensorcnt(0);
}

// ---------------- weight transpose + bf16 convert:  Wt[n*K+k] = W[k*N+n] ----------------
__global__ void transpose_bf16_kernel(const float* __restrict__ W, unsigned short* __restrict__ Wt,
                                      int K, int N) {
    int idx = blockIdx.x * 256 + threadIdx.x;
    if (idx < K * N) {
        int n = idx / K, k = idx - n * K;
        Wt[idx] = f2bf(W[(size_t)k * N + n]);
    }
}

__global__ void pack3_kernel(const float* __restrict__ a, const float* __restrict__ b,
                             const float* __restrict__ c, float* __restrict__ out, int n) {
    int i = blockIdx.x * 256 + threadIdx.x;
    if (i < n) out[i] = a[i];
    else if (i < 2 * n) out[i] = b[i - n];
    else if (i < 3 * n) out[i] = c[i - 2 * n];
}

// ---------------- layernorm (f32 in -> bf16 out) ----------------
__global__ __launch_bounds__(256) void ln_kernel(const float* __restrict__ x,
                                                 const float* __restrict__ g,
                                                 const float* __restrict__ bta,
                                                 unsigned short* __restrict__ y, int E) {
    __shared__ float red[256];
    int tid = threadIdx.x;
    const float* xr = x + (size_t)blockIdx.x * E;
    float s = 0.f;
    for (int i = tid; i < E; i += 256) s += xr[i];
    red[tid] = s; __syncthreads();
    for (int o = 128; o > 0; o >>= 1) { if (tid < o) red[tid] += red[tid + o]; __syncthreads(); }
    float mean = red[0] * (1.0f / (float)E);
    __syncthreads();
    float v = 0.f;
    for (int i = tid; i < E; i += 256) { float d = xr[i] - mean; v += d * d; }
    red[tid] = v; __syncthreads();
    for (int o = 128; o > 0; o >>= 1) { if (tid < o) red[tid] += red[tid + o]; __syncthreads(); }
    float rstd = rsqrtf(red[0] * (1.0f / (float)E) + 1e-5f);
    unsigned short* yr = y + (size_t)blockIdx.x * E;
    for (int i = tid; i < E; i += 256)
        yr[i] = f2bf((xr[i] - mean) * rstd * g[i] + bta[i]);
}

// ---------------- bf16 WMMA GEMM with TDM double-buffered staging ----------------
// C = A[M,K] * Bt[N,K]^T + bias (+resid). act: 0 none, 1 exact GELU.
__global__ __launch_bounds__(128) void gemm_kernel(const unsigned short* __restrict__ A,
                                                   const unsigned short* __restrict__ Bt,
                                                   const float* __restrict__ bias,
                                                   const float* __restrict__ resid,
                                                   unsigned short* __restrict__ outb,
                                                   float* __restrict__ outf,
                                                   int M, int N, int K, int act) {
    __shared__ __align__(16) unsigned short sA[2][64 * 64];
    __shared__ __align__(16) unsigned short sB[2][64 * 64];
    int tid = threadIdx.x, wave = tid >> 5, lane = tid & 31;
    int wm = wave >> 1, wn = wave & 1;
    int m0 = blockIdx.x * 64, n0 = blockIdx.y * 64;

    f32x8 acc[2][2] = {};

    // prologue: DMA first K-tile of A and B into buffer 0 (wave 0 only; EXEC-independent DMA)
    if (tid < 32) {
        tdm_load_tile64(sA[0], A  + (size_t)m0 * K, (unsigned)K, (unsigned)M);
        tdm_load_tile64(sB[0], Bt + (size_t)n0 * K, (unsigned)K, (unsigned)N);
        tdm_wait0();
    }
    __syncthreads();

    int cur = 0;
    for (int k0 = 0; k0 < K; k0 += 64) {
        // issue DMA for next tile into the other buffer while we compute on this one
        if (k0 + 64 < K && tid < 32) {
            tdm_load_tile64(sA[cur ^ 1], A  + (size_t)m0 * K + k0 + 64, (unsigned)K, (unsigned)M);
            tdm_load_tile64(sB[cur ^ 1], Bt + (size_t)n0 * K + k0 + 64, (unsigned)K, (unsigned)N);
        }
#pragma unroll
        for (int ks = 0; ks < 64; ks += 32) {
            bf16x16 a0 = load_frag(&sA[cur][(wm * 32 + 0)  * 64 + ks], 64, lane);
            bf16x16 a1 = load_frag(&sA[cur][(wm * 32 + 16) * 64 + ks], 64, lane);
            bf16x16 b0 = load_frag(&sB[cur][(wn * 32 + 0)  * 64 + ks], 64, lane);
            bf16x16 b1 = load_frag(&sB[cur][(wn * 32 + 16) * 64 + ks], 64, lane);
            acc[0][0] = wmma_bf16(a0, b0, acc[0][0]);
            acc[0][1] = wmma_bf16(a0, b1, acc[0][1]);
            acc[1][0] = wmma_bf16(a1, b0, acc[1][0]);
            acc[1][1] = wmma_bf16(a1, b1, acc[1][1]);
        }
        if (k0 + 64 < K && tid < 32) tdm_wait0();   // next buffer complete
        __syncthreads();
        cur ^= 1;
    }

    int half = lane >> 4, nl = lane & 15;
#pragma unroll
    for (int ti = 0; ti < 2; ++ti)
#pragma unroll
        for (int tj = 0; tj < 2; ++tj)
#pragma unroll
            for (int r = 0; r < 8; ++r) {
                int row = m0 + wm * 32 + ti * 16 + half * 8 + r;
                int col = n0 + wn * 32 + tj * 16 + nl;
                float v = acc[ti][tj][r] + bias[col];
                if (act == 1) v = 0.5f * v * (1.0f + erff(v * 0.7071067811865475f));
                if (resid) v += resid[(size_t)row * N + col];
                if (outf) outf[(size_t)row * N + col] = v;
                if (outb) outb[(size_t)row * N + col] = f2bf(v);
            }
}

// ---------------- flash attention: qkv bf16 [B,S,3E] -> ctx bf16 [B,S,E] ----------------
__global__ __launch_bounds__(128) void attn_kernel(const unsigned short* __restrict__ qkv,
                                                   unsigned short* __restrict__ ctx) {
    __shared__ __align__(16) unsigned short sQ[64 * 64];
    __shared__ __align__(16) unsigned short sK[64 * 64];
    __shared__ __align__(16) unsigned short sV[64 * 64];   // transposed: [d][token]
    __shared__ __align__(16) unsigned short sP[64 * 64];
    __shared__ float rowM[64], rowL[64], pmax[2][64], psum[2][64];

    int tid = threadIdx.x, wave = tid >> 5, lane = tid & 31;
    int wm = wave >> 1, wn = wave & 1;
    int half = lane >> 4, nl = lane & 15;
    int q0 = blockIdx.x * 64;
    int b  = blockIdx.y / HQ, h = blockIdx.y % HQ;
    size_t baseTok = (size_t)b * SQ * E3;
    int qcol = h * DQ, kcol = EQ + h * DQ, vcol = 2 * EQ + h * DQ;
    const float rscale = 0.036084391824351615f;  // 1/sqrt(768)

    // Q tile via TDM (rows q0..q0+63, cols qcol..qcol+63, row stride E3)
    if (tid < 32) {
        tdm_load_tile64(sQ, qkv + baseTok + (size_t)q0 * E3 + qcol, (unsigned)E3, (unsigned)SQ);
    }
    if (tid < 64) { rowM[tid] = -INFINITY; rowL[tid] = 0.f; }
    if (tid < 32) tdm_wait0();
    __syncthreads();

    f32x8 accO[2][2] = {};

    for (int kt = 0; kt < SQ; kt += 64) {
        // K tile via TDM (token-major == N-major B layout); V transposed by all threads
        if (tid < 32)
            tdm_load_tile64(sK, qkv + baseTok + (size_t)kt * E3 + kcol, (unsigned)E3, (unsigned)SQ);
        for (int idx = tid; idx < 64 * 8; idx += 128) {
            int r = idx >> 3, c8 = (idx & 7) * 8;
            union { uint4 q; unsigned short s[8]; } u;
            u.q = *(const uint4*)&qkv[baseTok + (size_t)(kt + r) * E3 + vcol + c8];
#pragma unroll
            for (int j = 0; j < 8; ++j) sV[(c8 + j) * 64 + r] = u.s[j];
        }
        if (tid < 32) tdm_wait0();
        __syncthreads();

        // S = Q * K^T  (this wave's 32x32 subtile)
        f32x8 accS[2][2] = {};
#pragma unroll
        for (int ks = 0; ks < 64; ks += 32) {
            bf16x16 a0 = load_frag(&sQ[(wm * 32 + 0)  * 64 + ks], 64, lane);
            bf16x16 a1 = load_frag(&sQ[(wm * 32 + 16) * 64 + ks], 64, lane);
            bf16x16 b0 = load_frag(&sK[(wn * 32 + 0)  * 64 + ks], 64, lane);
            bf16x16 b1 = load_frag(&sK[(wn * 32 + 16) * 64 + ks], 64, lane);
            accS[0][0] = wmma_bf16(a0, b0, accS[0][0]);
            accS[0][1] = wmma_bf16(a0, b1, accS[0][1]);
            accS[1][0] = wmma_bf16(a1, b0, accS[1][0]);
            accS[1][1] = wmma_bf16(a1, b1, accS[1][1]);
        }
        float rmax[2][8];
#pragma unroll
        for (int ti = 0; ti < 2; ++ti)
#pragma unroll
            for (int r = 0; r < 8; ++r) {
                accS[ti][0][r] *= rscale;
                accS[ti][1][r] *= rscale;
                rmax[ti][r] = fmaxf(accS[ti][0][r], accS[ti][1][r]);
            }
        // cross-lane row max (masks 1..8 keep the two 16-lane halves separate)
#pragma unroll
        for (int m = 1; m < 16; m <<= 1)
#pragma unroll
            for (int ti = 0; ti < 2; ++ti)
#pragma unroll
                for (int r = 0; r < 8; ++r)
                    rmax[ti][r] = fmaxf(rmax[ti][r], __shfl_xor(rmax[ti][r], m, 32));
        if (nl == 0) {
#pragma unroll
            for (int ti = 0; ti < 2; ++ti)
#pragma unroll
                for (int r = 0; r < 8; ++r)
                    pmax[wn][wm * 32 + ti * 16 + half * 8 + r] = rmax[ti][r];
        }
        __syncthreads();

        // P = exp(S - m_new), rescale O, accumulate row sums, spill P to LDS
        float rsum[2][8];
#pragma unroll
        for (int ti = 0; ti < 2; ++ti)
#pragma unroll
            for (int r = 0; r < 8; ++r) {
                int row = wm * 32 + ti * 16 + half * 8 + r;
                float mo = rowM[row];
                float mn = fmaxf(mo, fmaxf(pmax[0][row], pmax[1][row]));
                float p0 = __expf(accS[ti][0][r] - mn);
                float p1 = __expf(accS[ti][1][r] - mn);
                rsum[ti][r] = p0 + p1;
                float f = __expf(mo - mn);
                accO[ti][0][r] *= f;
                accO[ti][1][r] *= f;
                sP[row * 64 + wn * 32 + nl]      = f2bf(p0);
                sP[row * 64 + wn * 32 + 16 + nl] = f2bf(p1);
            }
#pragma unroll
        for (int m = 1; m < 16; m <<= 1)
#pragma unroll
            for (int ti = 0; ti < 2; ++ti)
#pragma unroll
                for (int r = 0; r < 8; ++r)
                    rsum[ti][r] += __shfl_xor(rsum[ti][r], m, 32);
        if (nl == 0) {
#pragma unroll
            for (int ti = 0; ti < 2; ++ti)
#pragma unroll
                for (int r = 0; r < 8; ++r)
                    psum[wn][wm * 32 + ti * 16 + half * 8 + r] = rsum[ti][r];
        }
        __syncthreads();

        // running stats update (reads old rowM; nobody else touches it here)
        if (tid < 64) {
            float mo = rowM[tid];
            float mn = fmaxf(mo, fmaxf(pmax[0][tid], pmax[1][tid]));
            rowL[tid] = rowL[tid] * __expf(mo - mn) + psum[0][tid] + psum[1][tid];
            rowM[tid] = mn;
        }

        // O += P * V  (N = d, Kdim = token; sV is [d][token] i.e. N-major)
#pragma unroll
        for (int ks = 0; ks < 64; ks += 32) {
            bf16x16 a0 = load_frag(&sP[(wm * 32 + 0)  * 64 + ks], 64, lane);
            bf16x16 a1 = load_frag(&sP[(wm * 32 + 16) * 64 + ks], 64, lane);
            bf16x16 b0 = load_frag(&sV[(wn * 32 + 0)  * 64 + ks], 64, lane);
            bf16x16 b1 = load_frag(&sV[(wn * 32 + 16) * 64 + ks], 64, lane);
            accO[0][0] = wmma_bf16(a0, b0, accO[0][0]);
            accO[0][1] = wmma_bf16(a0, b1, accO[0][1]);
            accO[1][0] = wmma_bf16(a1, b0, accO[1][0]);
            accO[1][1] = wmma_bf16(a1, b1, accO[1][1]);
        }
        __syncthreads();
    }

    // epilogue: O /= rowL ; ctx[b, q0+row, h*64 + d]
#pragma unroll
    for (int ti = 0; ti < 2; ++ti)
#pragma unroll
        for (int tj = 0; tj < 2; ++tj)
#pragma unroll
            for (int r = 0; r < 8; ++r) {
                int row = wm * 32 + ti * 16 + half * 8 + r;
                int d   = wn * 32 + tj * 16 + nl;
                float o = accO[ti][tj][r] / rowL[row];
                ctx[((size_t)b * SQ + q0 + row) * EQ + h * DQ + d] = f2bf(o);
            }
}

// ---------------- host launcher ----------------
extern "C" void kernel_launch(void* const* d_in, const int* in_sizes, int n_in,
                              void* d_out, int out_size, void* d_ws, size_t ws_size,
                              hipStream_t stream) {
    const float* x     = (const float*)d_in[0];
    const float* ln1_g = (const float*)d_in[1];
    const float* ln1_b = (const float*)d_in[2];
    const float* wq    = (const float*)d_in[3];
    const float* bq    = (const float*)d_in[4];
    const float* wk    = (const float*)d_in[5];
    const float* bk    = (const float*)d_in[6];
    const float* wv    = (const float*)d_in[7];
    const float* bv    = (const float*)d_in[8];
    const float* wo    = (const float*)d_in[9];
    const float* bo    = (const float*)d_in[10];
    const float* w1    = (const float*)d_in[11];
    const float* b1    = (const float*)d_in[12];
    const float* w2    = (const float*)d_in[13];
    const float* b2    = (const float*)d_in[14];
    const float* ln2_g = (const float*)d_in[15];
    const float* ln2_b = (const float*)d_in[16];
    float* out = (float*)d_out;

    const int M = BQ * SQ;   // 4096 token rows
    char* ws = (char*)d_ws;
    size_t off = 0;
    auto take = [&](size_t bytes) -> void* {
        void* p = ws + off;
        off += (bytes + 255) & ~(size_t)255;
        return p;
    };
    unsigned short* wqkv_t = (unsigned short*)take((size_t)E3 * EQ * 2);
    unsigned short* wo_t   = (unsigned short*)take((size_t)EQ * EQ * 2);
    unsigned short* w1_t   = (unsigned short*)take((size_t)E4 * EQ * 2);
    unsigned short* w2_t   = (unsigned short*)take((size_t)EQ * E4 * 2);
    float*          bqkv   = (float*)take((size_t)E3 * 4);
    unsigned short* y1     = (unsigned short*)take((size_t)M * EQ * 2);
    unsigned short* qkv    = (unsigned short*)take((size_t)M * E3 * 2);
    unsigned short* ctx    = (unsigned short*)take((size_t)M * EQ * 2);
    float*          y2     = (float*)take((size_t)M * EQ * 4);
    unsigned short* z2     = (unsigned short*)take((size_t)M * EQ * 2);
    unsigned short* h1     = (unsigned short*)take((size_t)M * E4 * 2);

    // weights -> bf16, transposed to [N][K]
    int nEE = EQ * EQ, nE4 = EQ * E4;
    transpose_bf16_kernel<<<(nEE + 255) / 256, 256, 0, stream>>>(wq, wqkv_t,                   EQ, EQ);
    transpose_bf16_kernel<<<(nEE + 255) / 256, 256, 0, stream>>>(wk, wqkv_t + (size_t)EQ*EQ,   EQ, EQ);
    transpose_bf16_kernel<<<(nEE + 255) / 256, 256, 0, stream>>>(wv, wqkv_t + (size_t)2*EQ*EQ, EQ, EQ);
    transpose_bf16_kernel<<<(nEE + 255) / 256, 256, 0, stream>>>(wo, wo_t, EQ, EQ);
    transpose_bf16_kernel<<<(nE4 + 255) / 256, 256, 0, stream>>>(w1, w1_t, EQ, E4);
    transpose_bf16_kernel<<<(nE4 + 255) / 256, 256, 0, stream>>>(w2, w2_t, E4, EQ);
    pack3_kernel<<<(E3 + 255) / 256, 256, 0, stream>>>(bq, bk, bv, bqkv, EQ);

    // LN1
    ln_kernel<<<M, 256, 0, stream>>>(x, ln1_g, ln1_b, y1, EQ);

    // fused QKV projection: [4096,768] x [768,2304]
    gemm_kernel<<<dim3(M / 64, E3 / 64), 128, 0, stream>>>(
        y1, wqkv_t, bqkv, nullptr, qkv, nullptr, M, E3, EQ, 0);

    // attention
    attn_kernel<<<dim3(SQ / 64, BQ * HQ), 128, 0, stream>>>(qkv, ctx);

    // output projection + residual: y2 = x + ctx@wo + bo
    gemm_kernel<<<dim3(M / 64, EQ / 64), 128, 0, stream>>>(
        ctx, wo_t, bo, x, nullptr, y2, M, EQ, EQ, 0);

    // LN2
    ln_kernel<<<M, 256, 0, stream>>>(y2, ln2_g, ln2_b, z2, EQ);

    // MLP: h1 = gelu(z2@w1 + b1)
    gemm_kernel<<<dim3(M / 64, E4 / 64), 128, 0, stream>>>(
        z2, w1_t, b1, nullptr, h1, nullptr, M, E4, EQ, 1);

    // out = y2 + h1@w2 + b2
    gemm_kernel<<<dim3(M / 64, EQ / 64), 128, 0, stream>>>(
        h1, w2_t, b2, y2, nullptr, out, M, EQ, E4, 0);
}
